// DreamsEncoder_90245852824252
// MI455X (gfx1250) — matrix operand order repaired and checked by reading, not compiled, not run on hardware
//
#include <hip/hip_runtime.h>
#include <hip/hip_bf16.h>
#include <stdint.h>

// ---------------------------------------------------------------------------
// DREAMS-style encoder for MI455X (gfx1250).
// All GEMMs on v_wmma_f32_16x16x32_bf16, fp32 accumulate.
// 128x128 block tile, 8 waves * (32x64) -> 8 WMMA per k-step per wave.
// Weights pre-transposed to bf16 [K][N] once per launch; W tiles staged to
// LDS via global_load_async_to_lds_b128 (ASYNCcnt).
// ---------------------------------------------------------------------------

typedef __attribute__((ext_vector_type(16))) __bf16 v16bf;
typedef __attribute__((ext_vector_type(8)))  float  v8f;
typedef int v4i __attribute__((vector_size(16)));

union FragB16 {
    v16bf v;
    __bf16 h[16];
    uint4  q2[2];
};

union Pack8 {
    uint4  q;
    __bf16 h[8];
};

#define TOK    4096      // B*N
#define DMODEL 1024
#define NHEAD  8
#define HDIM   128
#define SEQ    256
#define NBATCH 16

#define BM 128
#define BN 128
#define BK 32

// ---------------- async global->LDS (CDNA5) ----------------
#if defined(__HIP_DEVICE_COMPILE__) && __has_builtin(__builtin_amdgcn_global_load_async_to_lds_b128)
#define USE_ASYNC_LDS 1
__device__ __forceinline__ void async_cp16(void* l, const void* g) {
    __builtin_amdgcn_global_load_async_to_lds_b128(
        (__attribute__((address_space(1))) v4i*)(g),
        (__attribute__((address_space(3))) v4i*)(l),
        0, 0);
}
__device__ __forceinline__ void async_wait0() {
#if __has_builtin(__builtin_amdgcn_s_wait_asynccnt)
    __builtin_amdgcn_s_wait_asynccnt(0);
#else
    asm volatile("s_wait_asynccnt 0" ::: "memory");
#endif
}
#else
#define USE_ASYNC_LDS 0
__device__ __forceinline__ void async_cp16(void* l, const void* g) {
    *(uint4*)l = *(const uint4*)g;
}
__device__ __forceinline__ void async_wait0() {}
#endif

__device__ __forceinline__ uint4 ld_g16(const __bf16* p) { return *(const uint4*)p; }

// ===========================================================================
// Weight transpose + bf16 convert:  W (N x K, f32)  ->  Wt (Kpad x ldout, bf16)
// Zero-padded outside [K, N].
// ===========================================================================
__global__ __launch_bounds__(256) void k_transpose_bf16(
    const float* __restrict__ W, int N, int K,
    __bf16* __restrict__ Wt, int ldout, int Kpad)
{
    __shared__ __bf16 tile[32][33];
    const int k0 = blockIdx.x * 32;
    const int n0 = blockIdx.y * 32;
    const int t  = threadIdx.x;
    const int r  = t >> 3;          // 0..31
    const int c4 = (t & 7) * 4;     // 0,4,..,28

    const int n  = n0 + r;
    const int nc = n < N ? n : N - 1;
#pragma unroll
    for (int j = 0; j < 4; ++j) {
        const int k  = k0 + c4 + j;
        const int kc = k < K ? k : K - 1;
        float v = W[(long)nc * K + kc];
        if (n >= N || k >= K) v = 0.0f;
        tile[c4 + j][r] = (__bf16)v;
    }
    __syncthreads();
    const int k = k0 + r;
    if (k < Kpad) {
#pragma unroll
        for (int j = 0; j < 4; ++j) {
            const int n2 = n0 + c4 + j;
            if (n2 < ldout) Wt[(long)k * ldout + n2] = tile[r][c4 + j];
        }
    }
}

// ===========================================================================
// Generic NT GEMM: C[M,N] = act(A[M,K] @ Wt[K][N] + bias + resid)
// A: f32 (M % 128 == 0, K % 32 == 0);  Wt: bf16 pre-transposed, ldWt >= N.
// 128x128 tile, K-step 32, 256 threads / 8 waves (4x2), 8 WMMA per wave/step.
// ===========================================================================
__global__ __launch_bounds__(256) void k_gemm(
    const float* __restrict__ A, int ldA,
    const __bf16* __restrict__ Wt, int ldWt,
    const float* __restrict__ bias,
    const float* __restrict__ resid, int ldR,
    float* __restrict__ C, int ldC,
    int M, int N, int K, int relu)
{
    __shared__ __bf16 sA[BM][BK];      // [m][k]
    __shared__ __bf16 sB[BK][BN];      // [k][n]

    const int tid  = threadIdx.x;
    const int lane = tid & 31;
    const int w    = tid >> 5;
    const int wr   = w >> 1;           // 0..3 (32 rows each)
    const int wc   = w & 1;            // 0..1 (64 cols each)
    const int tM   = blockIdx.y * BM;
    const int tN   = blockIdx.x * BN;

    const int la_r = tid >> 1;         // 0..127
    const int la_k = (tid & 1) * 16;   // 0 or 16
    const int lb_k = tid >> 3;         // 0..31
    const int lb_n = (tid & 7) * 16;   // 0..112

    const float* aptr  = A  + (long)(tM + la_r) * ldA + la_k;
    const __bf16* bptr = Wt + (long)lb_k * ldWt + tN + lb_n;

    v8f acc[2][4] = {};

    for (int k0 = 0; k0 < K; k0 += BK) {
        // ---- stage A tile: 4x b128 load -> pk_bf16 -> 2x b128 LDS store ----
        {
            const float* ap = aptr + k0;
            __builtin_prefetch(ap + BK, 0, 3);
            const float4 f0 = *(const float4*)(ap);
            const float4 f1 = *(const float4*)(ap + 4);
            const float4 f2 = *(const float4*)(ap + 8);
            const float4 f3 = *(const float4*)(ap + 12);
            Pack8 u0, u1;
            u0.h[0] = (__bf16)f0.x; u0.h[1] = (__bf16)f0.y;
            u0.h[2] = (__bf16)f0.z; u0.h[3] = (__bf16)f0.w;
            u0.h[4] = (__bf16)f1.x; u0.h[5] = (__bf16)f1.y;
            u0.h[6] = (__bf16)f1.z; u0.h[7] = (__bf16)f1.w;
            u1.h[0] = (__bf16)f2.x; u1.h[1] = (__bf16)f2.y;
            u1.h[2] = (__bf16)f2.z; u1.h[3] = (__bf16)f2.w;
            u1.h[4] = (__bf16)f3.x; u1.h[5] = (__bf16)f3.y;
            u1.h[6] = (__bf16)f3.z; u1.h[7] = (__bf16)f3.w;
            *(uint4*)&sA[la_r][la_k]     = u0.q;
            *(uint4*)&sA[la_r][la_k + 8] = u1.q;
        }
        // ---- stage W tile: 2x async 16B global->LDS copies ----
        {
            const __bf16* bp = bptr + (long)k0 * ldWt;
            async_cp16(&sB[lb_k][lb_n],     bp);
            async_cp16(&sB[lb_k][lb_n + 8], bp + 8);
        }
        async_wait0();
        __syncthreads();

        FragB16 af[2];
#pragma unroll
        for (int rr = 0; rr < 2; ++rr) {
            const int m  = wr * 32 + rr * 16 + (lane & 15);
            const int kb = (lane >> 4) * 8;
            af[rr].q2[0] = *(const uint4*)&sA[m][kb];
            af[rr].q2[1] = *(const uint4*)&sA[m][kb + 16];
        }
#pragma unroll
        for (int cc = 0; cc < 4; ++cc) {
            FragB16 bfr;
            const int n0 = wc * 64 + cc * 16;
#pragma unroll
            for (int j = 0; j < 8; ++j) {
                bfr.h[j]     = sB[(lane >> 4) * 8 + j][n0 + (lane & 15)];
                bfr.h[j + 8] = sB[16 + (lane >> 4) * 8 + j][n0 + (lane & 15)];
            }
#pragma unroll
            for (int rr = 0; rr < 2; ++rr)
                acc[rr][cc] = __builtin_amdgcn_wmma_f32_16x16x32_bf16(
                    false, af[rr].v, false, bfr.v, (short)0, acc[rr][cc], false, false);
        }
        __syncthreads();
    }

    // ---- epilogue: bias / residual / relu, fp32 store ----
#pragma unroll
    for (int rr = 0; rr < 2; ++rr)
#pragma unroll
        for (int cc = 0; cc < 4; ++cc) {
            const int col = tN + wc * 64 + cc * 16 + (lane & 15);
            const int cbc = col < N ? col : 0;
            const float bv = bias ? bias[cbc] : 0.0f;
#pragma unroll
            for (int vr = 0; vr < 8; ++vr) {
                const int row = tM + wr * 32 + rr * 16 + vr + ((lane >> 4) * 8);
                float v = acc[rr][cc][vr] + bv;
                if (resid) v += resid[(long)row * ldR + cbc];
                if (relu)  v = fmaxf(v, 0.0f);
                if (col < N) C[(long)row * ldC + col] = v;
            }
        }
}

// ===========================================================================
// Fourier GEMM: A[m,k] = cos/sin(2*pi*mz[m]*fb[k mod NF]) generated on the fly.
// C = relu(A @ Wt + bias).  Same 128x128 tiling; K loop to Kpad (zero-padded).
// ===========================================================================
__global__ __launch_bounds__(256) void k_gemm_fourier(
    const float* __restrict__ spec,
    const float* __restrict__ fb, int NF,
    const __bf16* __restrict__ Wt, int ldWt,
    const float* __restrict__ bias,
    float* __restrict__ C, int ldC,
    int N, int K, int Kpad)
{
    __shared__ __bf16 sA[BM][BK];
    __shared__ __bf16 sB[BK][BN];

    const int tid  = threadIdx.x;
    const int lane = tid & 31;
    const int w    = tid >> 5;
    const int wr   = w >> 1;
    const int wc   = w & 1;
    const int tM   = blockIdx.y * BM;
    const int tN   = blockIdx.x * BN;

    const int la_r = tid >> 1;
    const int la_k = (tid & 1) * 16;
    const int lb_k = tid >> 3;
    const int lb_n = (tid & 7) * 16;

    const float TWO_PI = 6.28318530717958647692f;
    const float mz = spec[(tM + la_r) * 2 + 0];
    const __bf16* bptr = Wt + (long)lb_k * ldWt + tN + lb_n;

    v8f acc[2][4] = {};

    for (int k0 = 0; k0 < Kpad; k0 += BK) {
        // ---- synthesize A tile (branch-free: clamped index + selects) ----
        {
            Pack8 u0, u1;
#pragma unroll
            for (int j = 0; j < 16; ++j) {
                const int kk     = k0 + la_k + j;
                const bool issin = kk >= NF;
                int ki = issin ? kk - NF : kk;
                ki = ki < NF ? ki : NF - 1;
                const float f   = fb[ki];
                const float ang = TWO_PI * mz * f;
                float v = issin ? __sinf(ang) : __cosf(ang);
                if (kk >= K) v = 0.0f;
                if (j < 8) u0.h[j] = (__bf16)v;
                else       u1.h[j - 8] = (__bf16)v;
            }
            *(uint4*)&sA[la_r][la_k]     = u0.q;
            *(uint4*)&sA[la_r][la_k + 8] = u1.q;
        }
        {
            const __bf16* bp = bptr + (long)k0 * ldWt;
            async_cp16(&sB[lb_k][lb_n],     bp);
            async_cp16(&sB[lb_k][lb_n + 8], bp + 8);
        }
        async_wait0();
        __syncthreads();

        FragB16 af[2];
#pragma unroll
        for (int rr = 0; rr < 2; ++rr) {
            const int m  = wr * 32 + rr * 16 + (lane & 15);
            const int kb = (lane >> 4) * 8;
            af[rr].q2[0] = *(const uint4*)&sA[m][kb];
            af[rr].q2[1] = *(const uint4*)&sA[m][kb + 16];
        }
#pragma unroll
        for (int cc = 0; cc < 4; ++cc) {
            FragB16 bfr;
            const int n0 = wc * 64 + cc * 16;
#pragma unroll
            for (int j = 0; j < 8; ++j) {
                bfr.h[j]     = sB[(lane >> 4) * 8 + j][n0 + (lane & 15)];
                bfr.h[j + 8] = sB[16 + (lane >> 4) * 8 + j][n0 + (lane & 15)];
            }
#pragma unroll
            for (int rr = 0; rr < 2; ++rr)
                acc[rr][cc] = __builtin_amdgcn_wmma_f32_16x16x32_bf16(
                    false, af[rr].v, false, bfr.v, (short)0, acc[rr][cc], false, false);
        }
        __syncthreads();
    }

#pragma unroll
    for (int rr = 0; rr < 2; ++rr)
#pragma unroll
        for (int cc = 0; cc < 4; ++cc) {
            const int col = tN + wc * 64 + cc * 16 + (lane & 15);
            const int cbc = col < N ? col : 0;
            const float bv = bias[cbc];
#pragma unroll
            for (int vr = 0; vr < 8; ++vr) {
                const int row = tM + wr * 32 + rr * 16 + vr + ((lane >> 4) * 8);
                const float v = fmaxf(acc[rr][cc][vr] + bv, 0.0f);
                if (col < N) C[(long)row * ldC + col] = v;
            }
        }
}

// ===========================================================================
// Repack qkv (token, 3D) f32 -> bf16  Q (bh,n,d), K^T (bh,d,n), V (bh,n,d)
// ===========================================================================
__global__ __launch_bounds__(256) void k_qkv_pack(
    const float* __restrict__ qkv,
    __bf16* __restrict__ qb, __bf16* __restrict__ kt, __bf16* __restrict__ vb)
{
    const int idx = blockIdx.x * 256 + threadIdx.x;   // TOK*1024
    const int tok = idx >> 10;
    const int hd  = idx & 1023;
    const int b   = tok >> 8;
    const int n   = tok & 255;
    const int h   = hd >> 7;
    const int d   = hd & 127;
    const int bh  = b * NHEAD + h;
    const float q = qkv[(long)tok * 3072 + hd];
    const float k = qkv[(long)tok * 3072 + 1024 + hd];
    const float v = qkv[(long)tok * 3072 + 2048 + hd];
    qb[((long)bh * SEQ + n) * HDIM + d] = (__bf16)q;
    kt[((long)bh * HDIM + d) * SEQ + n] = (__bf16)k;
    vb[((long)bh * SEQ + n) * HDIM + d] = (__bf16)v;
}

// ===========================================================================
// Fused attention, one workgroup per (32-query tile, head, batch).
// Fragments loaded directly from packed bf16 global buffers (aligned 16B).
// ===========================================================================
__global__ __launch_bounds__(256) void k_attn(
    const __bf16* __restrict__ qb,
    const __bf16* __restrict__ kt,
    const __bf16* __restrict__ vb,
    const float* __restrict__ spec,
    float* __restrict__ o)
{
    const int qt = blockIdx.x;
    const int h  = blockIdx.y;
    const int b  = blockIdx.z;
    const int bh = b * NHEAD + h;

    __shared__ float sc[32][256];
    __shared__ float kmz[256];
    __shared__ float qmz[32];
    __shared__ int   qmask[32];
    __shared__ float rpart[32][8];
    __shared__ float rinv[32];

    const int tid  = threadIdx.x;
    const int lane = tid & 31;
    const int w    = tid >> 5;
    const float scale = 0.0883883476483184f;   // 128^-0.5

    kmz[tid] = spec[(b * SEQ + tid) * 2 + 0];
    if (tid < 32) {
        const int tq = b * SEQ + qt * 32 + tid;
        qmz[tid]   = spec[tq * 2 + 0];
        qmask[tid] = (spec[tq * 2 + 1] == 0.0f) ? 1 : 0;
    }
    __syncthreads();

    // ---------------- phase 1: S = Q K^T ----------------
    v8f sfr[2][2] = {{{}, {}}, {{}, {}}};
    const __bf16* qbase = qb + ((long)bh * SEQ + qt * 32) * HDIM;
    const __bf16* kbase = kt + (long)bh * HDIM * SEQ;
#pragma unroll
    for (int ds = 0; ds < 4; ++ds) {
        FragB16 af[2];
#pragma unroll
        for (int rt = 0; rt < 2; ++rt) {
            const __bf16* p = qbase + (long)(rt * 16 + (lane & 15)) * HDIM
                              + ds * 32 + ((lane >> 4) * 8);
            af[rt].q2[0] = ld_g16(p);
            af[rt].q2[1] = ld_g16(p + 16);
        }
#pragma unroll
        for (int ci = 0; ci < 2; ++ci) {
            FragB16 bf;
            const __bf16* p = kbase + (long)(ds * 32 + ((lane >> 4) * 8)) * SEQ + (2 * w + ci) * 16 + (lane & 15);
#pragma unroll
            for (int j = 0; j < 8; ++j) {
                bf.h[j]     = p[(long)j * SEQ];
                bf.h[j + 8] = p[(long)(16 + j) * SEQ];
            }
#pragma unroll
            for (int rt = 0; rt < 2; ++rt)
                sfr[rt][ci] = __builtin_amdgcn_wmma_f32_16x16x32_bf16(
                    false, af[rt].v, false, bf.v, (short)0, sfr[rt][ci], false, false);
        }
    }

    // scores -> LDS with scale + mz bias + query mask
#pragma unroll
    for (int rt = 0; rt < 2; ++rt)
#pragma unroll
        for (int ci = 0; ci < 2; ++ci)
#pragma unroll
            for (int vr = 0; vr < 8; ++vr) {
                const int q   = rt * 16 + vr + ((lane >> 4) * 8);
                const int key = (2 * w + ci) * 16 + (lane & 15);
                float s = sfr[rt][ci][vr] * scale + qmz[q] - kmz[key];
                if (qmask[q]) s = -1.0e9f;
                sc[q][key] = s;
            }
    __syncthreads();

    // ---------------- phase 2: softmax over 256 keys ----------------
    {
        const int r   = tid >> 3;
        const int seg = tid & 7;
        float mx = -3.4e38f;
        for (int c = seg * 32; c < seg * 32 + 32; ++c) mx = fmaxf(mx, sc[r][c]);
        rpart[r][seg] = mx;
        __syncthreads();
        if (seg == 0) {
            float m = rpart[r][0];
#pragma unroll
            for (int j = 1; j < 8; ++j) m = fmaxf(m, rpart[r][j]);
            rinv[r] = m;               // temporarily row max
        }
        __syncthreads();
        const float m = rinv[r];
        float s = 0.0f;
        for (int c = seg * 32; c < seg * 32 + 32; ++c) {
            const float e = __expf(sc[r][c] - m);
            sc[r][c] = e;
            s += e;
        }
        __syncthreads();
        rpart[r][seg] = s;
        __syncthreads();
        if (seg == 0) {
            float t = 0.0f;
#pragma unroll
            for (int j = 0; j < 8; ++j) t += rpart[r][j];
            rinv[r] = 1.0f / t;
        }
        __syncthreads();
    }

    // ---------------- phase 3: O = P V ----------------
    v8f oa[2] = {{}, {}};
    const int dt = w;
    const __bf16* vbase = vb + (long)bh * SEQ * HDIM;
#pragma unroll
    for (int ks = 0; ks < 8; ++ks) {
        FragB16 bf;
        const __bf16* p = vbase + (long)(ks * 32 + ((lane >> 4) * 8)) * HDIM + dt * 16 + (lane & 15);
#pragma unroll
        for (int j = 0; j < 8; ++j) {
            bf.h[j]     = p[(long)j * HDIM];
            bf.h[j + 8] = p[(long)(16 + j) * HDIM];
        }
#pragma unroll
        for (int rt = 0; rt < 2; ++rt) {
            FragB16 af;
            const int q  = rt * 16 + (lane & 15);
            const int kb = (lane >> 4) * 8;
            const float ri = rinv[q];
#pragma unroll
            for (int j = 0; j < 8; ++j) {
                af.h[j]     = (__bf16)(sc[q][ks * 32 + kb + j] * ri);
                af.h[j + 8] = (__bf16)(sc[q][ks * 32 + kb + 16 + j] * ri);
            }
            oa[rt] = __builtin_amdgcn_wmma_f32_16x16x32_bf16(
                false, af.v, false, bf.v, (short)0, oa[rt], false, false);
        }
    }

#pragma unroll
    for (int rt = 0; rt < 2; ++rt)
#pragma unroll
        for (int vr = 0; vr < 8; ++vr) {
            const int q   = qt * 32 + rt * 16 + vr + ((lane >> 4) * 8);
            const int col = h * HDIM + dt * 16 + (lane & 15);
            o[(long)(b * SEQ + q) * DMODEL + col] = oa[rt][vr];
        }
}

// ===========================================================================
// LayerNorm (one workgroup per row of D=1024)
// ===========================================================================
__global__ __launch_bounds__(256) void k_ln(
    const float* __restrict__ x,
    const float* __restrict__ g,
    const float* __restrict__ bta,
    float* __restrict__ y, int D)
{
    __shared__ float red[256];
    const int tid = threadIdx.x;
    const float* xr = x + (long)blockIdx.x * D;
    float* yr = y + (long)blockIdx.x * D;

    float s = 0.0f;
    for (int i = tid; i < D; i += 256) s += xr[i];
    red[tid] = s; __syncthreads();
    for (int off = 128; off; off >>= 1) {
        if (tid < off) red[tid] += red[tid + off];
        __syncthreads();
    }
    const float mean = red[0] / (float)D;
    __syncthreads();

    float v = 0.0f;
    for (int i = tid; i < D; i += 256) { const float d = xr[i] - mean; v += d * d; }
    red[tid] = v; __syncthreads();
    for (int off = 128; off; off >>= 1) {
        if (tid < off) red[tid] += red[tid + off];
        __syncthreads();
    }
    const float rstd = rsqrtf(red[0] / (float)D + 1e-5f);
    __syncthreads();

    for (int i = tid; i < D; i += 256)
        yr[i] = (xr[i] - mean) * rstd * g[i] + bta[i];
}

// ===========================================================================
// Peak embedding: x[:, 980 + j] = relu(spec . pw[j] + pb[j])
// ===========================================================================
__global__ __launch_bounds__(256) void k_peak(
    const float* __restrict__ spec,
    const float* __restrict__ pw,
    const float* __restrict__ pb,
    float* __restrict__ x)
{
    const int idx = blockIdx.x * 256 + threadIdx.x;
    if (idx >= TOK * 44) return;
    const int t = idx / 44;
    const int j = idx - t * 44;
    const float v = spec[t * 2] * pw[j * 2] + spec[t * 2 + 1] * pw[j * 2 + 1] + pb[j];
    x[(long)t * DMODEL + 980 + j] = fmaxf(v, 0.0f);
}

// ===========================================================================
// Host orchestration
// ===========================================================================
extern "C" void kernel_launch(void* const* d_in, const int* in_sizes, int n_in,
                              void* d_out, int out_size, void* d_ws, size_t ws_size,
                              hipStream_t stream)
{
    const float* spec = (const float*)d_in[0];
    const float* fbv  = (const float*)d_in[1];
    const float* pw   = (const float*)d_in[2];
    const float* pb   = (const float*)d_in[3];
    const float* fw[5] = { (const float*)d_in[4], (const float*)d_in[6],
                           (const float*)d_in[8], (const float*)d_in[10],
                           (const float*)d_in[12] };
    const float* fbias[5] = { (const float*)d_in[5], (const float*)d_in[7],
                              (const float*)d_in[9], (const float*)d_in[11],
                              (const float*)d_in[13] };
    const float* w_qkvo = (const float*)d_in[14];
    const float* ffn_w1 = (const float*)d_in[15];
    const float* ffn_w2 = (const float*)d_in[16];
    const float* ln1_g  = (const float*)d_in[17];
    const float* ln1_b  = (const float*)d_in[18];
    const float* ln2_g  = (const float*)d_in[19];
    const float* ln2_b  = (const float*)d_in[20];
    const float* lnf_g  = (const float*)d_in[21];
    const float* lnf_b  = (const float*)d_in[22];

    const int NF  = in_sizes[1];
    const int K0  = 2 * NF;
    const int Kp0 = ((K0 + 31) / 32) * 32;

    // ---- f32 workspace ----
    float* x    = (float*)d_ws;
    float* y    = x    + (long)TOK * DMODEL;
    float* qkv  = y    + (long)TOK * DMODEL;
    float* obuf = qkv  + (long)TOK * 3 * DMODEL;
    float* h1   = obuf + (long)TOK * DMODEL;
    float* hm0  = h1   + (long)TOK * 4096;
    float* hm1  = hm0  + (long)TOK * 512;

    // ---- bf16 workspace (64B aligned) ----
    __bf16* bp = (__bf16*)((((uintptr_t)(hm1 + (long)TOK * 512)) + 63) & ~(uintptr_t)63);
    __bf16* qbuf = bp;  bp += (long)TOK * DMODEL;
    __bf16* ktb  = bp;  bp += (long)TOK * DMODEL;
    __bf16* vbuf = bp;  bp += (long)TOK * DMODEL;
    __bf16* fw0t = bp;  bp += (long)Kp0 * 512;
    __bf16* fw1t = bp;  bp += 512L * 512;
    __bf16* fw2t = bp;  bp += 512L * 512;
    __bf16* fw3t = bp;  bp += 512L * 512;
    __bf16* fw4t = bp;  bp += 512L * 1024;
    __bf16 *qkvt[7], *wot[7], *w1t[7], *w2t[7];
    for (int l = 0; l < 7; ++l) {
        qkvt[l] = bp; bp += 1024L * 3072;
        wot[l]  = bp; bp += 1024L * 1024;
        w1t[l]  = bp; bp += 1024L * 4096;
        w2t[l]  = bp; bp += 4096L * 1024;
    }

    const dim3 blk(256);
    auto tpose = [&](const float* W, int N, int K, __bf16* Wt, int ldout, int Kpad) {
        k_transpose_bf16<<<dim3(Kpad / 32, ldout / 32), blk, 0, stream>>>(
            W, N, K, Wt, ldout, Kpad);
    };

    // ---- transpose + bf16-convert all weights (once per launch) ----
    tpose(fw[0], 512, K0, fw0t, 512, Kp0);
    tpose(fw[1], 512, 512, fw1t, 512, 512);
    tpose(fw[2], 512, 512, fw2t, 512, 512);
    tpose(fw[3], 512, 512, fw3t, 512, 512);
    tpose(fw[4], 980, 512, fw4t, 1024, 512);
    for (int l = 0; l < 7; ++l) {
        const float* wq = w_qkvo + (long)l * 4 * DMODEL * DMODEL;
        tpose(wq,                          3072, 1024, qkvt[l], 3072, 1024);
        tpose(wq + 3L * DMODEL * DMODEL,   1024, 1024, wot[l],  1024, 1024);
        tpose(ffn_w1 + (long)l * 4096 * DMODEL, 4096, 1024, w1t[l], 4096, 1024);
        tpose(ffn_w2 + (long)l * DMODEL * 4096, 1024, 4096, w2t[l], 1024, 4096);
    }

    // ---- Fourier MLP front-end (M=4096 rows -> 32 row-tiles of 128) ----
    k_gemm_fourier<<<dim3(4, TOK / BM), blk, 0, stream>>>(
        spec, fbv, NF, fw0t, 512, fbias[0], hm0, 512, 512, K0, Kp0);
    k_gemm<<<dim3(4, TOK / BM), blk, 0, stream>>>(
        hm0, 512, fw1t, 512, fbias[1], nullptr, 0, hm1, 512, TOK, 512, 512, 1);
    k_gemm<<<dim3(4, TOK / BM), blk, 0, stream>>>(
        hm1, 512, fw2t, 512, fbias[2], nullptr, 0, hm0, 512, TOK, 512, 512, 1);
    k_gemm<<<dim3(4, TOK / BM), blk, 0, stream>>>(
        hm0, 512, fw3t, 512, fbias[3], nullptr, 0, hm1, 512, TOK, 512, 512, 1);
    k_gemm<<<dim3(8, TOK / BM), blk, 0, stream>>>(
        hm1, 512, fw4t, 1024, fbias[4], nullptr, 0, x, DMODEL, TOK, 980, 512, 1);
    k_peak<<<dim3((TOK * 44 + 255) / 256), blk, 0, stream>>>(spec, pw, pb, x);

    // ---- 7 transformer layers ----
    for (int l = 0; l < 7; ++l) {
        k_ln<<<dim3(TOK), blk, 0, stream>>>(x, ln1_g + l * DMODEL, ln1_b + l * DMODEL, y, DMODEL);
        k_gemm<<<dim3(24, TOK / BM), blk, 0, stream>>>(
            y, DMODEL, qkvt[l], 3072, nullptr, nullptr, 0, qkv, 3 * DMODEL,
            TOK, 3 * DMODEL, DMODEL, 0);
        k_qkv_pack<<<dim3(TOK * DMODEL / 256), blk, 0, stream>>>(qkv, qbuf, ktb, vbuf);
        k_attn<<<dim3(8, NHEAD, NBATCH), blk, 0, stream>>>(qbuf, ktb, vbuf, spec, obuf);
        k_gemm<<<dim3(8, TOK / BM), blk, 0, stream>>>(
            obuf, DMODEL, wot[l], 1024, nullptr, x, DMODEL, x, DMODEL,
            TOK, DMODEL, DMODEL, 0);

        k_ln<<<dim3(TOK), blk, 0, stream>>>(x, ln2_g + l * DMODEL, ln2_b + l * DMODEL, y, DMODEL);
        k_gemm<<<dim3(32, TOK / BM), blk, 0, stream>>>(
            y, DMODEL, w1t[l], 4096, nullptr, nullptr, 0, h1, 4096,
            TOK, 4096, DMODEL, 1);
        k_gemm<<<dim3(8, TOK / BM), blk, 0, stream>>>(
            h1, 4096, w2t[l], 1024, nullptr, x, DMODEL, x, DMODEL,
            TOK, DMODEL, 4096, 0);
    }

    k_ln<<<dim3(TOK), blk, 0, stream>>>(x, lnf_g, lnf_b, (float*)d_out, DMODEL);
}
